// SequenceTransformerLinearProjector_81527069213222
// MI455X (gfx1250) — matrix operand order repaired and checked by reading, not compiled, not run on hardware
//
#include <hip/hip_runtime.h>
#include <hip/hip_bf16.h>

typedef __attribute__((ext_vector_type(16))) __bf16 v16bf;
typedef __attribute__((ext_vector_type(8)))  float  v8f;

#define NSEQ 8192
#define DM   512
#define FFD  2048
#define LLMD 4096

// ---------------- positional encoding add ----------------
__global__ __launch_bounds__(256) void pe_add_kernel(const float* __restrict__ xin,
                                                     float* __restrict__ xout) {
  int idx = blockIdx.x * 256 + threadIdx.x;            // NSEQ*DM divisible by 256
  int pos = idx >> 9;
  int c   = idx & 511;
  float f   = expf((float)(c & ~1) * (-9.210340371976184f / 512.0f)); // exp(2i * -ln(1e4)/D)
  float ang = (float)pos * f;
  float pe  = (c & 1) ? cosf(ang) : sinf(ang);
  xout[idx] = xin[idx] + pe;
}

// ---------------- layernorm: fp32 in -> bf16 out ----------------
__global__ __launch_bounds__(256) void ln_kernel(const float* __restrict__ x,
                                                 const float* __restrict__ g,
                                                 const float* __restrict__ b,
                                                 __bf16* __restrict__ out) {
  int wave = threadIdx.x >> 5, lane = threadIdx.x & 31;
  int row = blockIdx.x * 8 + wave;
  const float* xr = x + (size_t)row * DM;
  float v[16];
  float s = 0.f, s2 = 0.f;
#pragma unroll
  for (int i = 0; i < 16; ++i) {
    float t = xr[lane + i * 32];
    v[i] = t; s += t; s2 += t * t;
  }
#pragma unroll
  for (int m = 1; m < 32; m <<= 1) {
    s  += __shfl_xor(s,  m, 32);
    s2 += __shfl_xor(s2, m, 32);
  }
  float mean = s * (1.0f / 512.0f);
  float var  = s2 * (1.0f / 512.0f) - mean * mean;
  float rstd = rsqrtf(var + 1e-5f);
  __bf16* orow = out + (size_t)row * DM;
#pragma unroll
  for (int i = 0; i < 16; ++i) {
    int c = lane + i * 32;
    orow[c] = (__bf16)((v[i] - mean) * rstd * g[c] + b[c]);
  }
}

// ---------------- generic bf16 WMMA GEMM ----------------
// C[M,N] = A[M,K](bf16,row-major) x Bt[N,K](bf16, column-of-B contiguous in K)
// wave tile 32x64 (2 A-frags x 4 B-frags -> 8 v_wmma per K-step)
// block = 8 waves -> 256x64 block tile.
// epilogue: +bias, optional exact GELU, optional +resid, write f32 and/or bf16.
__global__ __launch_bounds__(256) void gemm_bf16_kernel(
    const __bf16* __restrict__ A, const __bf16* __restrict__ Bt,
    const float* __restrict__ bias, const float* __restrict__ resid,
    float* __restrict__ outF, __bf16* __restrict__ outB,
    int N, int K, int act) {
  int wave = threadIdx.x >> 5, lane = threadIdx.x & 31;
  int rowBase = blockIdx.y * 256 + wave * 32;
  int colBase = blockIdx.x * 64;

  const v8f vzero = {0.f,0.f,0.f,0.f,0.f,0.f,0.f,0.f};
  v8f acc[2][4];
#pragma unroll
  for (int r = 0; r < 2; ++r)
#pragma unroll
    for (int t = 0; t < 4; ++t) acc[r][t] = vzero;

  // A fragment (16x32 bf16): lane row = lane&15 ; K chunks {0..7,16..23} (lanes<16)
  //                          or {8..15,24..31} (lanes>=16)
  int am  = lane & 15;
  int akb = (lane >> 4) << 3;   // 0 or 8
  // B fragment (32x16 bf16): lane col = lane&15 ; K chunks same as A
  int bn  = lane & 15;
  int bkb = (lane >> 4) << 3;   // 0 or 8

  const __bf16* arow0 = A + (size_t)(rowBase + am) * K + akb;
  const __bf16* arow1 = arow0 + (size_t)16 * K;
  const __bf16* bp    = Bt + (size_t)(colBase + bn) * K + bkb;
  size_t bstride = (size_t)16 * K;

  for (int k0 = 0; k0 < K; k0 += 32) {
    // prefetch next A tiles (A streams from HBM; B weights stay hot in L2)
    if (k0 + 128 < K) {
      __builtin_prefetch(arow0 + k0 + 128, 0, 1);
      __builtin_prefetch(arow1 + k0 + 128, 0, 1);
    }
    v16bf a0, a1;
#pragma unroll
    for (int e = 0; e < 8; ++e)  { a0[e] = arow0[k0 + e];     a1[e] = arow1[k0 + e]; }
#pragma unroll
    for (int e = 8; e < 16; ++e) { a0[e] = arow0[k0 + 8 + e]; a1[e] = arow1[k0 + 8 + e]; }
    v16bf bb[4];
#pragma unroll
    for (int t = 0; t < 4; ++t) {
      const __bf16* btp = bp + (size_t)t * bstride + k0;
#pragma unroll
      for (int e = 0; e < 16; ++e) bb[t][e] = btp[e < 8 ? e : 8 + e];
    }
#pragma unroll
    for (int t = 0; t < 4; ++t) {
      acc[0][t] = __builtin_amdgcn_wmma_f32_16x16x32_bf16(
          false, a0, false, bb[t], (short)0, acc[0][t], false, false);
      acc[1][t] = __builtin_amdgcn_wmma_f32_16x16x32_bf16(
          false, a1, false, bb[t], (short)0, acc[1][t], false, false);
    }
  }

  int mh = lane >> 4;
#pragma unroll
  for (int r = 0; r < 2; ++r) {
#pragma unroll
    for (int t = 0; t < 4; ++t) {
      int col = colBase + t * 16 + (lane & 15);
      float bv = bias ? bias[col] : 0.f;
#pragma unroll
      for (int e = 0; e < 8; ++e) {
        int row = rowBase + r * 16 + e + 8 * mh;
        float vl = acc[r][t][e] + bv;
        if (act == 1) vl = 0.5f * vl * (1.0f + erff(vl * 0.70710678118654752f)); // exact GELU
        size_t oi = (size_t)row * N + col;
        if (resid) vl += resid[oi];
        if (outF) outF[oi] = vl;
        if (outB) outB[oi] = (__bf16)vl;
      }
    }
  }
}

// ---------------- q softmax over head-dim (64) + scale, out bf16 ----------------
__global__ __launch_bounds__(256) void qsm_kernel(const float* __restrict__ q,
                                                  __bf16* __restrict__ out) {
  int t = threadIdx.x;
  int row = blockIdx.x * 4 + (t >> 6);
  int sub = t & 63;                       // 8 threads per head, 8 elems per thread
  size_t base = (size_t)row * DM + sub * 8;
  float v[8];
  float m = -1e30f;
#pragma unroll
  for (int i = 0; i < 8; ++i) { v[i] = q[base + i]; m = fmaxf(m, v[i]); }
#pragma unroll
  for (int msk = 1; msk < 8; msk <<= 1) m = fmaxf(m, __shfl_xor(m, msk, 32));
  float s = 0.f;
#pragma unroll
  for (int i = 0; i < 8; ++i) { v[i] = expf(v[i] - m); s += v[i]; }
#pragma unroll
  for (int msk = 1; msk < 8; msk <<= 1) s += __shfl_xor(s, msk, 32);
  float inv = 0.125f / s;                 // scale = DH^-0.5 = 1/8
#pragma unroll
  for (int i = 0; i < 8; ++i) out[base + i] = (__bf16)(v[i] * inv);
}

// ---------------- k softmax over sequence dim (8192), in-place fp32 ----------------
__global__ __launch_bounds__(256) void ksm_kernel(float* __restrict__ k) {
  __shared__ float red[8][32];
  int t = threadIdx.x;
  int ci = t & 31, j = t >> 5;
  int col = blockIdx.x * 32 + ci;
  float m = -1e30f;
  for (int r = j; r < NSEQ; r += 8) m = fmaxf(m, k[(size_t)r * DM + col]);
  red[j][ci] = m; __syncthreads();
  if (j == 0) {
    float mm = red[0][ci];
    for (int q = 1; q < 8; ++q) mm = fmaxf(mm, red[q][ci]);
    red[0][ci] = mm;
  }
  __syncthreads();
  m = red[0][ci];
  __syncthreads();
  float s = 0.f;
  for (int r = j; r < NSEQ; r += 8) s += expf(k[(size_t)r * DM + col] - m);
  red[j][ci] = s; __syncthreads();
  if (j == 0) {
    float ss = 0.f;
    for (int q = 0; q < 8; ++q) ss += red[q][ci];
    red[0][ci] = ss;
  }
  __syncthreads();
  float inv = 1.0f / red[0][ci];
  for (int r = j; r < NSEQ; r += 8) {
    size_t idx = (size_t)r * DM + col;
    k[idx] = expf(k[idx] - m) * inv;
  }
}

// ---------------- ctx[h][d][e] = sum_n ksm[n,h*64+d] * v[n,h*64+e] ----------------
__global__ __launch_bounds__(256) void ctx_kernel(const float* __restrict__ kf,
                                                  const float* __restrict__ vf,
                                                  float* __restrict__ ctx) {
  __shared__ float ks[32][64];
  __shared__ float vs[32][64];
  int t = threadIdx.x;
  int head = blockIdx.y;
  int base0 = blockIdx.x * 512;
  int e  = t & 63;
  int d0 = (t >> 6) * 16;
  float acc[16];
#pragma unroll
  for (int i = 0; i < 16; ++i) acc[i] = 0.f;

  for (int base = base0; base < base0 + 512; base += 32) {
    for (int idx = t; idx < 2048; idx += 256) {
      int n = idx >> 6, c = idx & 63;
      size_t gi = (size_t)(base + n) * DM + head * 64 + c;
      ks[n][c] = kf[gi];
      vs[n][c] = vf[gi];
    }
    __syncthreads();
#pragma unroll 4
    for (int n = 0; n < 32; ++n) {
      float vv = vs[n][e];
#pragma unroll
      for (int j = 0; j < 16; ++j) acc[j] += ks[n][d0 + j] * vv;
    }
    __syncthreads();
  }
#pragma unroll
  for (int j = 0; j < 16; ++j)
    atomicAdd(&ctx[head * 4096 + (d0 + j) * 64 + e], acc[j]);
}

// ---------------- pack ctx into block-diagonal bf16 Bt[n][k] ----------------
__global__ __launch_bounds__(256) void pack_ctx_kernel(const float* __restrict__ ctx,
                                                       __bf16* __restrict__ out) {
  int idx = blockIdx.x * 256 + threadIdx.x;   // 512*512
  int n = idx >> 9, k = idx & 511;
  int hn = n >> 6, hk = k >> 6;
  out[idx] = (hn == hk) ? (__bf16)ctx[hn * 4096 + (k & 63) * 64 + (n & 63)]
                        : (__bf16)0.f;
}

// ---------------- misc small kernels ----------------
__global__ __launch_bounds__(256) void zero_f32_kernel(float* __restrict__ p, int n) {
  int idx = blockIdx.x * 256 + threadIdx.x;
  if (idx < n) p[idx] = 0.f;
}
__global__ __launch_bounds__(256) void cvt_f32_bf16_kernel(const float* __restrict__ in,
                                                           __bf16* __restrict__ out) {
  int idx = blockIdx.x * 256 + threadIdx.x;
  out[idx] = (__bf16)in[idx];
}
// weight [K,N] fp32 row-major -> Bt[N,K] bf16
__global__ __launch_bounds__(256) void tcvt_kernel(const float* __restrict__ in,
                                                   __bf16* __restrict__ out,
                                                   int K, int N) {
  int idx = blockIdx.x * 256 + threadIdx.x;   // K*N divisible by 256 here
  int n = idx / K, k = idx - n * K;
  out[idx] = (__bf16)in[(size_t)k * N + n];
}

// ---------------- host side ----------------
static inline void launch_gemm(const __bf16* A, const __bf16* Bt, const float* bias,
                               const float* resid, float* outF, __bf16* outB,
                               int M, int N, int K, int act, hipStream_t s) {
  dim3 g(N / 64, M / 256);
  gemm_bf16_kernel<<<g, 256, 0, s>>>(A, Bt, bias, resid, outF, outB, N, K, act);
}

extern "C" void kernel_launch(void* const* d_in, const int* in_sizes, int n_in,
                              void* d_out, int out_size, void* d_ws, size_t ws_size,
                              hipStream_t stream) {
  const float* x_in  = (const float*)d_in[0];
  const float* ln1_g = (const float*)d_in[1];
  const float* ln1_b = (const float*)d_in[2];
  const float* Wq    = (const float*)d_in[3];
  const float* Wk    = (const float*)d_in[4];
  const float* Wv    = (const float*)d_in[5];
  const float* Wo    = (const float*)d_in[6];
  const float* bo    = (const float*)d_in[7];
  const float* ln2_g = (const float*)d_in[8];
  const float* ln2_b = (const float*)d_in[9];
  const float* W1    = (const float*)d_in[10];
  const float* b1    = (const float*)d_in[11];
  const float* W2    = (const float*)d_in[12];
  const float* b2    = (const float*)d_in[13];
  const float* projW = (const float*)d_in[14];
  const float* projb = (const float*)d_in[15];

  char* ws = (char*)d_ws;
  size_t off = 0;
  auto alloc = [&](size_t bytes) -> char* {
    char* p = ws + off;
    off = (off + bytes + 255) & ~(size_t)255;
    return p;
  };

  // Long-lived buffers
  float*  xw  = (float*) alloc((size_t)NSEQ * DM * 4);     // residual stream (whole run)
  __bf16* hb  = (__bf16*)alloc((size_t)NSEQ * DM * 2);     // LN output
  float*  qf  = (float*) alloc((size_t)NSEQ * DM * 4);     // q pre-softmax (dead after qsm)
  float*  kf  = (float*) alloc((size_t)NSEQ * DM * 4);     // k / ksm (dead after ctx)
  float*  vf  = (float*) alloc((size_t)NSEQ * DM * 4);     // v        (dead after ctx)
  __bf16* qb  = (__bf16*)alloc((size_t)NSEQ * DM * 2);     // q softmax bf16
  float*  ctx = (float*) alloc((size_t)8 * 64 * 64 * 4);
  __bf16* cbd = (__bf16*)alloc((size_t)DM * DM * 2);
  // Aliased buffers (lifetimes disjoint with their hosts):
  __bf16* ob = (__bf16*)qf;                                   // 8MB, after qf is dead
  __bf16* xb = (__bf16*)((char*)qf + (size_t)NSEQ * DM * 2);  // 8MB, final stage only
  __bf16* f1 = (__bf16*)kf;                                   // 32MB over kf+vf, FF only

  __bf16* wqT[2], *wkT[2], *wvT[2], *woT[2], *w1T[2], *w2T[2];
  for (int l = 0; l < 2; ++l) {
    wqT[l] = (__bf16*)alloc((size_t)DM * DM * 2);
    wkT[l] = (__bf16*)alloc((size_t)DM * DM * 2);
    wvT[l] = (__bf16*)alloc((size_t)DM * DM * 2);
    woT[l] = (__bf16*)alloc((size_t)DM * DM * 2);
    w1T[l] = (__bf16*)alloc((size_t)DM * FFD * 2);
    w2T[l] = (__bf16*)alloc((size_t)FFD * DM * 2);
  }
  __bf16* prT = (__bf16*)alloc((size_t)DM * LLMD * 2);

  // ---- weight prep: fp32 [K,N] -> bf16 [N,K] ----
  const int bDD = (DM * DM) / 256;
  const int bDF = (DM * FFD) / 256;
  const int bDL = (DM * LLMD) / 256;
  for (int l = 0; l < 2; ++l) {
    size_t oDD = (size_t)l * DM * DM;
    size_t oDF = (size_t)l * DM * FFD;
    tcvt_kernel<<<bDD, 256, 0, stream>>>(Wq + oDD, wqT[l], DM, DM);
    tcvt_kernel<<<bDD, 256, 0, stream>>>(Wk + oDD, wkT[l], DM, DM);
    tcvt_kernel<<<bDD, 256, 0, stream>>>(Wv + oDD, wvT[l], DM, DM);
    tcvt_kernel<<<bDD, 256, 0, stream>>>(Wo + oDD, woT[l], DM, DM);
    tcvt_kernel<<<bDF, 256, 0, stream>>>(W1 + oDF, w1T[l], DM, FFD);
    tcvt_kernel<<<bDF, 256, 0, stream>>>(W2 + oDF, w2T[l], FFD, DM);
  }
  tcvt_kernel<<<bDL, 256, 0, stream>>>(projW, prT, DM, LLMD);

  // ---- x + positional encoding ----
  pe_add_kernel<<<(NSEQ * DM) / 256, 256, 0, stream>>>(x_in, xw);

  for (int l = 0; l < 2; ++l) {
    // --- attention block ---
    ln_kernel<<<NSEQ / 8, 256, 0, stream>>>(xw, ln1_g + l * DM, ln1_b + l * DM, hb);
    launch_gemm(hb, wqT[l], nullptr, nullptr, qf, nullptr, NSEQ, DM, DM, 0, stream);
    launch_gemm(hb, wkT[l], nullptr, nullptr, kf, nullptr, NSEQ, DM, DM, 0, stream);
    launch_gemm(hb, wvT[l], nullptr, nullptr, vf, nullptr, NSEQ, DM, DM, 0, stream);
    qsm_kernel<<<NSEQ / 4, 256, 0, stream>>>(qf, qb);      // qf dead after this
    ksm_kernel<<<DM / 32, 256, 0, stream>>>(kf);
    zero_f32_kernel<<<(8 * 64 * 64 + 255) / 256, 256, 0, stream>>>(ctx, 8 * 64 * 64);
    ctx_kernel<<<dim3(16, 8), 256, 0, stream>>>(kf, vf, ctx);  // kf,vf dead after this
    pack_ctx_kernel<<<(DM * DM) / 256, 256, 0, stream>>>(ctx, cbd);
    launch_gemm(qb, cbd, nullptr, nullptr, nullptr, ob, NSEQ, DM, DM, 0, stream);
    launch_gemm(ob, woT[l], bo + l * DM, xw, xw, nullptr, NSEQ, DM, DM, 0, stream);
    // --- feedforward block ---
    ln_kernel<<<NSEQ / 8, 256, 0, stream>>>(xw, ln2_g + l * DM, ln2_b + l * DM, hb);
    launch_gemm(hb, w1T[l], b1 + l * FFD, nullptr, nullptr, f1, NSEQ, FFD, DM, 1, stream);
    launch_gemm(f1, w2T[l], b2 + l * DM, xw, xw, nullptr, NSEQ, DM, FFD, 0, stream);
  }

  // ---- final projector ----
  cvt_f32_bf16_kernel<<<(NSEQ * DM) / 256, 256, 0, stream>>>(xw, xb);
  launch_gemm(xb, prT, projb, nullptr, (float*)d_out, nullptr, NSEQ, LLMD, DM, 0, stream);
}
